// BaselineGIN_64811056497271
// MI455X (gfx1250) — matrix operand order repaired and checked by reading, not compiled, not run on hardware
//
#include <hip/hip_runtime.h>
#include <hip/hip_bf16.h>

// ---------------------------------------------------------------------------
// GIN forward for MI455X (gfx1250, wave32, WMMA).
// Layers are L2-bandwidth bound (x fits in 192MB L2); node-MLP GEMMs use
// v_wmma_f32_16x16x32_bf16 so they are effectively free next to the scatter.
// ---------------------------------------------------------------------------

#define N_NODES  100000
#define N_EDGES  1600000
#define CH       64
#define OUT_C    8
#define N_LAYERS 3
#define N_GRAPHS 64
#define BN_EPS   1e-5f

typedef __attribute__((ext_vector_type(8)))  float  v8f;
typedef __attribute__((ext_vector_type(16))) __bf16 v16bf;
typedef __attribute__((ext_vector_type(8)))  __bf16 v8bf;

// --------------------------------------------------------------------------
__global__ void gin_zero(float* __restrict__ p, int n) {
    int i = blockIdx.x * blockDim.x + threadIdx.x;
    if (i < n) p[i] = 0.0f;
}

// --------------------------------------------------------------------------
// Edge scatter-add: agg[dst[e]][:] += x[src[e]][:].  16 threads per edge,
// float4 gather (L2-resident) + 4 global_atomic_add_f32.
__global__ void gin_scatter(const float* __restrict__ x,
                            const int*   __restrict__ src,
                            const int*   __restrict__ dst,
                            float*       __restrict__ agg) {
    long long t = (long long)blockIdx.x * blockDim.x + threadIdx.x;
    int e  = (int)(t >> 4);
    if (e >= N_EDGES) return;
    int c4 = (int)(t & 15) * 4;
    int s = src[e];
    int d = dst[e];
    const float4 val = *(const float4*)(x + (size_t)s * CH + c4);
    float* ap = agg + (size_t)d * CH + c4;
    atomicAdd(ap + 0, val.x);
    atomicAdd(ap + 1, val.y);
    atomicAdd(ap + 2, val.z);
    atomicAdd(ap + 3, val.w);
}

// --------------------------------------------------------------------------
// Fused node MLP: out = relu( W2^T( relu( BN( W1^T(x+agg) + b1 ) ) ) + b2 )
// Block: 256 threads (8 wave32), 128 nodes per block, bf16 WMMA 16x16x32.
__launch_bounds__(256, 2)
__global__ void gin_mlp(const float* __restrict__ x_in,
                        const float* __restrict__ agg,
                        const float* __restrict__ W1,
                        const float* __restrict__ b1,
                        const float* __restrict__ gam,
                        const float* __restrict__ bet,
                        const float* __restrict__ bmean,
                        const float* __restrict__ bvar,
                        const float* __restrict__ W2,
                        const float* __restrict__ b2,
                        float*       __restrict__ x_out) {
    __shared__ __bf16 ldsA[128][CH];   // A tile, later reused as hidden tile
    __shared__ __bf16 ldsB1[CH][CH];   // W1 transposed [n][k]
    __shared__ __bf16 ldsB2[CH][CH];   // W2 transposed [n][k]
    __shared__ float  bnScale[CH];     // gamma * rsqrt(var+eps)
    __shared__ float  bnShift[CH];     // beta + scale*(b1 - mean)
    __shared__ float  bias2[CH];

    const int tid   = threadIdx.x;
    const int lane  = tid & 31;
    const int wave  = tid >> 5;          // 0..7 -> M tile
    const int node0 = blockIdx.x * 128;

    // ---- stage weights (transposed -> contiguous K per output channel) ----
    for (int i = tid; i < CH * CH; i += 256) {
        int k = i >> 6, n = i & 63;
        ldsB1[n][k] = (__bf16)W1[i];
        ldsB2[n][k] = (__bf16)W2[i];
    }
    if (tid < CH) {
        float sc = gam[tid] * rsqrtf(bvar[tid] + BN_EPS);
        bnScale[tid] = sc;
        bnShift[tid] = bet[tid] + sc * (b1[tid] - bmean[tid]);
        bias2[tid]   = b2[tid];
    }
    // ---- stage A = bf16(x + agg), zero-padded past N_NODES ----
    for (int i = tid; i < 128 * CH / 2; i += 256) {
        int r = i >> 5;
        int c = (i & 31) * 2;
        int node = node0 + r;
        float a0 = 0.0f, a1 = 0.0f;
        if (node < N_NODES) {
            const float2 xv = *(const float2*)(x_in + (size_t)node * CH + c);
            const float2 av = *(const float2*)(agg  + (size_t)node * CH + c);
            a0 = xv.x + av.x;
            a1 = xv.y + av.y;
        }
        ldsA[r][c]     = (__bf16)a0;
        ldsA[r][c + 1] = (__bf16)a1;
    }
    __syncthreads();

    const int mrow  = lane & 15;   // row within 16x16 tile (A side)
    const int khalf = lane >> 4;   // K-half select per ISA fragment layout
    const v8f vzero = {};

    // =================== GEMM 1: (x+agg) @ W1 ===================
    v8f acc[4];
    acc[0] = vzero; acc[1] = vzero; acc[2] = vzero; acc[3] = vzero;

    #pragma unroll
    for (int k0 = 0; k0 < CH; k0 += 32) {
        const int am = wave * 16 + mrow;
        const v8bf alo = *(const v8bf*)&ldsA[am][k0 + khalf * 8];
        const v8bf ahi = *(const v8bf*)&ldsA[am][k0 + khalf * 8 + 16];
        v16bf afrag;
        #pragma unroll
        for (int i = 0; i < 8; ++i) { afrag[i] = alo[i]; afrag[i + 8] = ahi[i]; }

        #pragma unroll
        for (int nt = 0; nt < 4; ++nt) {
            const int bn = nt * 16 + mrow;
            const v8bf blo = *(const v8bf*)&ldsB1[bn][k0 + khalf * 8];
            const v8bf bhi = *(const v8bf*)&ldsB1[bn][k0 + khalf * 8 + 16];
            v16bf bfrag;
            #pragma unroll
            for (int i = 0; i < 8; ++i) { bfrag[i] = blo[i]; bfrag[i + 8] = bhi[i]; }
            acc[nt] = __builtin_amdgcn_wmma_f32_16x16x32_bf16(
                false, afrag, false, bfrag, (short)0, acc[nt], false, false);
        }
    }

    __syncthreads();   // all A-tile reads done before we overwrite with hidden

    // ---- epilogue 1: fused (b1 + BN) affine, ReLU, restage hidden (bf16) ---
    {
        const int rbase = wave * 16 + ((lane < 16) ? 0 : 8);
        #pragma unroll
        for (int nt = 0; nt < 4; ++nt) {
            const int n  = nt * 16 + mrow;
            const float sc = bnScale[n];
            const float sh = bnShift[n];
            #pragma unroll
            for (int r = 0; r < 8; ++r) {
                float h = fmaxf(sc * acc[nt][r] + sh, 0.0f);
                ldsA[rbase + r][n] = (__bf16)h;
            }
        }
    }
    __syncthreads();

    // =================== GEMM 2: hidden @ W2 ===================
    acc[0] = vzero; acc[1] = vzero; acc[2] = vzero; acc[3] = vzero;

    #pragma unroll
    for (int k0 = 0; k0 < CH; k0 += 32) {
        const int am = wave * 16 + mrow;
        const v8bf alo = *(const v8bf*)&ldsA[am][k0 + khalf * 8];
        const v8bf ahi = *(const v8bf*)&ldsA[am][k0 + khalf * 8 + 16];
        v16bf afrag;
        #pragma unroll
        for (int i = 0; i < 8; ++i) { afrag[i] = alo[i]; afrag[i + 8] = ahi[i]; }

        #pragma unroll
        for (int nt = 0; nt < 4; ++nt) {
            const int bn = nt * 16 + mrow;
            const v8bf blo = *(const v8bf*)&ldsB2[bn][k0 + khalf * 8];
            const v8bf bhi = *(const v8bf*)&ldsB2[bn][k0 + khalf * 8 + 16];
            v16bf bfrag;
            #pragma unroll
            for (int i = 0; i < 8; ++i) { bfrag[i] = blo[i]; bfrag[i + 8] = bhi[i]; }
            acc[nt] = __builtin_amdgcn_wmma_f32_16x16x32_bf16(
                false, afrag, false, bfrag, (short)0, acc[nt], false, false);
        }
    }

    // ---- epilogue 2: + b2, outer ReLU, store f32 ----
    {
        const int rbase = wave * 16 + ((lane < 16) ? 0 : 8);
        #pragma unroll
        for (int nt = 0; nt < 4; ++nt) {
            const int n  = nt * 16 + mrow;
            const float bb = bias2[n];
            #pragma unroll
            for (int r = 0; r < 8; ++r) {
                const int node = node0 + rbase + r;
                if (node < N_NODES) {
                    x_out[(size_t)node * CH + n] = fmaxf(acc[nt][r] + bb, 0.0f);
                }
            }
        }
    }
}

// --------------------------------------------------------------------------
__global__ void gin_pool(const float* __restrict__ x,
                         const int*   __restrict__ batch,
                         float*       __restrict__ pooled) {
    long long t = (long long)blockIdx.x * blockDim.x + threadIdx.x;
    if (t >= (long long)N_NODES * CH) return;
    int n = (int)(t >> 6);
    int c = (int)(t & 63);
    atomicAdd(&pooled[(size_t)batch[n] * CH + c], x[t]);
}

// --------------------------------------------------------------------------
__global__ void gin_head(const float* __restrict__ pooled,
                         const float* __restrict__ W1,
                         const float* __restrict__ b1,
                         const float* __restrict__ W2,
                         const float* __restrict__ b2,
                         float*       __restrict__ out) {
    __shared__ float h[N_GRAPHS * CH];
    const int i = threadIdx.x;           // graph id, 64 threads
    for (int n = 0; n < CH; ++n) {
        float acc = b1[n];
        for (int k = 0; k < CH; ++k) acc += pooled[i * CH + k] * W1[k * CH + n];
        h[i * CH + n] = fmaxf(acc, 0.0f);
    }
    __syncthreads();
    for (int o = 0; o < OUT_C; ++o) {
        float acc = b2[o];
        for (int k = 0; k < CH; ++k) acc += h[i * CH + k] * W2[k * OUT_C + o];
        out[i * OUT_C + o] = acc;
    }
}

// --------------------------------------------------------------------------
extern "C" void kernel_launch(void* const* d_in, const int* in_sizes, int n_in,
                              void* d_out, int out_size, void* d_ws, size_t ws_size,
                              hipStream_t stream) {
    (void)in_sizes; (void)n_in; (void)out_size; (void)ws_size;

    const float* x     = (const float*)d_in[0];
    const int*   ei    = (const int*)d_in[1];        // (2, N_EDGES)
    const int*   src   = ei;
    const int*   dst   = ei + N_EDGES;
    const int*   batch = (const int*)d_in[2];

    // params flattened in dict order: l{0,1,2}: W1,b1,g,be,m,v,W2,b2 ; head: W1,b1,W2,b2
    const float* P[28];
    for (int i = 0; i < 28; ++i) P[i] = (const float*)d_in[3 + i];

    float* x_cur  = (float*)d_ws;                         // 100000*64 f32
    float* aggbuf = x_cur + (size_t)N_NODES * CH;         // 100000*64 f32
    float* pooled = aggbuf + (size_t)N_NODES * CH;        // 64*64 f32

    const int zgrid_agg  = (N_NODES * CH + 255) / 256;
    const int sgrid      = (int)(((long long)N_EDGES * 16 + 255) / 256);
    const int mgrid      = (N_NODES + 127) / 128;

    for (int l = 0; l < N_LAYERS; ++l) {
        const float* xin = (l == 0) ? x : x_cur;
        const float** p  = P + l * 8;
        gin_zero<<<zgrid_agg, 256, 0, stream>>>(aggbuf, N_NODES * CH);
        gin_scatter<<<sgrid, 256, 0, stream>>>(xin, src, dst, aggbuf);
        gin_mlp<<<mgrid, 256, 0, stream>>>(xin, aggbuf,
                                           p[0], p[1], p[2], p[3], p[4], p[5], p[6], p[7],
                                           x_cur);
    }

    gin_zero<<<(N_GRAPHS * CH + 255) / 256, 256, 0, stream>>>(pooled, N_GRAPHS * CH);
    gin_pool<<<(int)(((long long)N_NODES * CH + 255) / 256), 256, 0, stream>>>(x_cur, batch, pooled);
    gin_head<<<1, N_GRAPHS, 0, stream>>>(pooled, P[24], P[25], P[26], P[27], (float*)d_out);
}
